// OmniRowRNN_41429254537239
// MI455X (gfx1250) — matrix operand order repaired and checked by reading, not compiled, not run on hardware
//
#include <hip/hip_runtime.h>
#include <hip/hip_bf16.h>

typedef unsigned short u16;
typedef __attribute__((ext_vector_type(16))) __bf16 v16bf;
typedef __attribute__((ext_vector_type(8)))  float  v8f;

union FragAB { v16bf v; unsigned u[8]; };
union FragC  { v8f   v; float    f[8]; };

__device__ __forceinline__ u16 f2bf(float f) {
  unsigned u = __float_as_uint(f);
  unsigned r = u + 0x7FFFu + ((u >> 16) & 1u);
  return (u16)(r >> 16);
}
__device__ __forceinline__ float bf2f(u16 h) {
  return __uint_as_float(((unsigned)h) << 16);
}

// A fragment: 16x32 bf16. lane(0..15)=row M (lanes 16..31 same rows, other K half).
// VGPR v<4: K = half*8 + 2v ; v>=4: K = 16 + half*8 + 2(v-4). 16B-contiguous halves.
__device__ __forceinline__ void loadA(FragAB& fr, const u16* base, int stride,
                                      int m0, int k0, int lane) {
  const u16* p = base + (m0 + (lane & 15)) * stride;
  const int h8 = (lane >> 4) << 3;
#pragma unroll
  for (int v = 0; v < 8; ++v) {
    int k = k0 + h8 + ((v < 4) ? (2 * v) : (16 + 2 * (v - 4)));
    fr.u[v] = *(const unsigned*)(p + k);
  }
}
// B fragment: 32x16 bf16 (KxN). lane = N col, K layout same as A (v<4: half*8+2v; v>=4: 16+half*8+2(v-4)).
// Source stored row-major as W[n][k] (k contiguous) == B(k,n).
__device__ __forceinline__ void loadB(FragAB& fr, const u16* base, int stride,
                                      int n0, int k0, int lane) {
  const u16* p = base + (n0 + (lane & 15)) * stride + k0 + ((lane >> 4) << 3);
#pragma unroll
  for (int v = 0; v < 8; ++v) fr.u[v] = *(const unsigned*)(p + ((v < 4) ? (2 * v) : (16 + 2 * (v - 4))));
}
__device__ __forceinline__ v8f bwmma(const FragAB& a, const FragAB& b, v8f c) {
  return __builtin_amdgcn_wmma_f32_16x16x32_bf16(false, a.v, false, b.v,
                                                 (short)0, c, false, false);
}
__device__ __forceinline__ void zeroC(FragC& c) {
#pragma unroll
  for (int j = 0; j < 8; ++j) c.f[j] = 0.f;
}
__device__ __forceinline__ float sigm(float x) {
  return 1.f / (1.f + __expf(-x));
}

// ---------------- prep kernels: fp32 -> padded bf16 layouts ----------------
__global__ void conv_kpad(u16* dst, const float* src, int R, int Ks, int Kd) {
  long long tot = (long long)R * Kd;
  for (long long i = (long long)blockIdx.x * blockDim.x + threadIdx.x; i < tot;
       i += (long long)gridDim.x * blockDim.x) {
    int k = (int)(i % Kd);
    long long r = i / Kd;
    dst[i] = (k < Ks) ? f2bf(src[r * Ks + k]) : (u16)0;
  }
}
__global__ void conv_rowpad(u16* dst, const float* src, int D, int Rs, int Rd, int K) {
  long long tot = (long long)D * Rd * K;
  for (long long i = (long long)blockIdx.x * blockDim.x + threadIdx.x; i < tot;
       i += (long long)gridDim.x * blockDim.x) {
    int k = (int)(i % K);
    long long t = i / K;
    int r = (int)(t % Rd);
    int d = (int)(t / Rd);
    dst[i] = (r < Rs) ? f2bf(src[((long long)d * Rs + r) * K + k]) : (u16)0;
  }
}
// pad h2e_b (4,129) -> f32 (4,160) with zeros; removes divergent guard in row phase
__global__ void pad_hb(float* dst, const float* src) {
  int i = blockIdx.x * blockDim.x + threadIdx.x;
  if (i < 4 * 160) {
    int n = i % 160, d = i / 160;
    dst[i] = (n < 129) ? src[d * 129 + n] : 0.f;
  }
}
// inp_b[d][r][c][b][160] with flips applied; k=128 is row feature, 129..159 = 0
__global__ void embed_kernel(u16* dst, const int* x, const float* emb) {
  const long long tot = 4LL * 32 * 32 * 32 * 160;
  for (long long i = (long long)blockIdx.x * blockDim.x + threadIdx.x; i < tot;
       i += (long long)gridDim.x * blockDim.x) {
    int k = (int)(i % 160);
    long long t = i / 160;
    int b = (int)(t & 31); t >>= 5;
    int c = (int)(t & 31); t >>= 5;
    int r = (int)(t & 31);
    int d = (int)(t >> 5);
    int rr = (d >> 1) ? 31 - r : r;
    int cc = (d & 1) ? 31 - c : c;
    float val;
    if (k < 128) {
      int vb = x[(b * 32 + rr) * 32 + cc];       // x[b,0,rr,cc]
      val = emb[((d * 258 + vb) << 7) + k];
    } else if (k == 128) {
      val = (float)rr * (2.0f / 32.0f) - 1.0f;
    } else {
      val = 0.f;
    }
    dst[i] = f2bf(val);
  }
}

// ---------------- persistent per-direction GRU kernel ----------------
// 16 waves: wave>>3 -> batch tile mt (0/16), wave&7 -> one 16-wide h-group.
// Per wave, 4 register WMMA accumulators: s_r/s_z accumulate gi+gh fused
// (only the sum is used by the sigmoid), a_n/h_n kept split for r*hn.
// Layer input for l>=1 is read directly from the hidden-state write buffer.
__global__ __launch_bounds__(512) void rnn_kernel(
    const u16* __restrict__ inp_b, const u16* __restrict__ wi0b,
    const u16* __restrict__ wirb, const u16* __restrict__ whhb,
    const u16* __restrict__ hwb, const float* __restrict__ b_ih,
    const float* __restrict__ b_hh, const float* __restrict__ hbp,
    u16* __restrict__ comb, u16* __restrict__ prevh, u16* __restrict__ out4h) {
  const int d = blockIdx.x;
  const int tid = threadIdx.x;
  const int wave = tid >> 5, lane = tid & 31;
  const int flip_r = d >> 1, flip_c = d & 1;
  const int mt = (wave >> 3) << 4;       // batch tile base (0 or 16)
  const int ntr = (wave & 7) << 4;       // owned h-group: r-gate col tile base
  const int h = ntr + (lane & 15);       // this lane's hidden column

  __shared__ __align__(16) u16 hsb[2][3 * 32 * 128];  // hidden state, dbl buf (48 KB)

  const u16* wi0 = wi0b + d * 384 * 160;
  const u16* wir = wirb + d * 2 * 384 * 128;
  const u16* whhD = whhb + d * 3 * 384 * 128;
  const u16* hw = hwb + d * 160 * 128;
  const float* hb = hbp + d * 160;           // zero-padded to 160
  u16* combd = comb + d * 32 * 32 * 160;     // [c][b][160]
  u16* prevd = prevh + d * 1024 * 128;       // [(c*32+b)][128]

  // preload per-lane gate biases into registers: rz pre-summed, n kept split
  float Brz[3], Bzz[3], Bin[3], Bhn[3];
#pragma unroll
  for (int l = 0; l < 3; ++l) {
    const float* bi = b_ih + (d * 3 + l) * 384;
    const float* bh = b_hh + (d * 3 + l) * 384;
    Brz[l] = bi[h] + bh[h];
    Bzz[l] = bi[128 + h] + bh[128 + h];
    Bin[l] = bi[256 + h];
    Bhn[l] = bh[256 + h];
  }

  const uint4 z4 = make_uint4(0u, 0u, 0u, 0u);
  {
    uint4* pz = (uint4*)prevd;                 // 1024*128 u16 = 16384 uint4
    for (int i = tid; i < 16384; i += 512) pz[i] = z4;
  }
  __threadfence_block();
  __syncthreads();

  for (int r = 0; r < 32; ++r) {
    const int rr = flip_r ? 31 - r : r;
    // ---- row phase: comb = inp_row + prev_h @ hw^T + hb ----
    // M = 1024 (c,b), N = 160 (padded 129), K = 128. 640 tiles, 40 per wave.
    const u16* inrow = inp_b + (((long long)d * 32 + r) * 32) * 32 * 160;
    for (int i = 0; i < 40; ++i) {
      int t = wave + i * 16;
      int mt2 = (t & 63) << 4;
      int nt2 = (t >> 6) << 4;
      FragC acc; zeroC(acc);
      for (int kc = 0; kc < 4; ++kc) {
        FragAB a, b;
        loadA(a, prevd, 128, mt2, kc * 32, lane);
        loadB(b, hw, 128, nt2, kc * 32, lane);
        acc.v = bwmma(a, b, acc.v);
      }
      int n = nt2 + (lane & 15);
      float hbv = hb[n];   // zero-padded: branch-free
#pragma unroll
      for (int j = 0; j < 8; ++j) {
        int m = mt2 + j + ((lane >> 4) << 3);
        float val = acc.f[j] + hbv + bf2f(inrow[m * 160 + n]);
        combd[m * 160 + n] = f2bf(val);
      }
    }
    {
      uint4* hz = (uint4*)&hsb[0][0];          // 3*32*128 u16 = 1536 uint4
      for (int i = tid; i < 1536; i += 512) hz[i] = z4;   // hs reset per row
    }
    __threadfence_block();
    __syncthreads();

    for (int c = 0; c < 32; ++c) {
      const int ccn = flip_c ? 31 - c : c;
      const u16* hs_rd = hsb[c & 1];
      u16* hs_wr = hsb[(c + 1) & 1];
      const u16* combc = combd + c * 32 * 160;

#pragma unroll
      for (int l = 0; l < 3; ++l) {
        const u16* Ax = (l == 0) ? combc : (hs_wr + (l - 1) * 32 * 128);
        const int As = (l == 0) ? 160 : 128;
        const int kcx = (l == 0) ? 5 : 4;
        const u16* Wx = (l == 0) ? wi0 : (wir + (l - 1) * 384 * 128);
        const int Ws = (l == 0) ? 160 : 128;
        const u16* Wh = whhD + l * 384 * 128;
        const u16* Hrd = hs_rd + l * 32 * 128;
        u16* Hwr = hs_wr + l * 32 * 128;

        FragC s_r, s_z, a_n, h_n;
        zeroC(s_r); zeroC(s_z); zeroC(a_n); zeroC(h_n);
#pragma unroll
        for (int kc = 0; kc < kcx; ++kc) {
          FragAB a, b;
          loadA(a, Ax, As, mt, kc * 32, lane);
          loadB(b, Wx, Ws, ntr, kc * 32, lane);       s_r.v = bwmma(a, b, s_r.v);
          loadB(b, Wx, Ws, 128 + ntr, kc * 32, lane); s_z.v = bwmma(a, b, s_z.v);
          loadB(b, Wx, Ws, 256 + ntr, kc * 32, lane); a_n.v = bwmma(a, b, a_n.v);
        }
#pragma unroll
        for (int kc = 0; kc < 4; ++kc) {
          FragAB a, b;
          loadA(a, Hrd, 128, mt, kc * 32, lane);
          loadB(b, Wh, 128, ntr, kc * 32, lane);       s_r.v = bwmma(a, b, s_r.v);
          loadB(b, Wh, 128, 128 + ntr, kc * 32, lane); s_z.v = bwmma(a, b, s_z.v);
          loadB(b, Wh, 128, 256 + ntr, kc * 32, lane); h_n.v = bwmma(a, b, h_n.v);
        }
        // GRU gates directly on accumulators: element (m,h) aligned across all 4
#pragma unroll
        for (int j = 0; j < 8; ++j) {
          const int m = mt + j + ((lane >> 4) << 3);
          float rg = sigm(s_r.f[j] + Brz[l]);
          float zg = sigm(s_z.f[j] + Bzz[l]);
          float ng = tanhf(a_n.f[j] + Bin[l] + rg * (h_n.f[j] + Bhn[l]));
          float hp = bf2f(Hrd[m * 128 + h]);
          float hv = (1.f - zg) * ng + zg * hp;
          u16 hbf = f2bf(hv);
          Hwr[m * 128 + h] = hbf;
          if (l == 2) {
            prevd[(c * 32 + m) * 128 + h] = hbf;
            out4h[(((m * 32 + rr) * 32 + ccn) << 9) + (d << 7) + h] = hbf;
          }
        }
        __syncthreads();   // one barrier per layer: publishes hs_wr slices
      }
    }
    __threadfence_block();
    __syncthreads();
  }
}

// ---------------- final projection: (32768 x 512) @ out_w^T (+bias) ----------------
__global__ __launch_bounds__(256) void proj_kernel(const u16* __restrict__ A,
                                                   const u16* __restrict__ Bw,
                                                   const float* __restrict__ bias,
                                                   float* __restrict__ out) {
  const int lane = threadIdx.x & 31, wave = threadIdx.x >> 5;
  const int mt = blockIdx.x << 4;
  const int ntile = blockIdx.y * 8 + wave;
  if (ntile >= 17) return;  // wave-uniform
  const int nt = ntile << 4;
  FragC acc; zeroC(acc);
#pragma unroll 4
  for (int kc = 0; kc < 16; ++kc) {
    FragAB a, b;
    loadA(a, A, 512, mt, kc * 32, lane);
    loadB(b, Bw, 512, nt, kc * 32, lane);
    acc.v = bwmma(a, b, acc.v);
  }
  int v = nt + (lane & 15);
  if (v < 258) {
    float bv = bias[v];
#pragma unroll
    for (int j = 0; j < 8; ++j) {
      int m = mt + j + ((lane >> 4) << 3);
      out[m * 258 + v] = acc.f[j] + bv;
    }
  }
}

extern "C" void kernel_launch(void* const* d_in, const int* in_sizes, int n_in,
                              void* d_out, int out_size, void* d_ws, size_t ws_size,
                              hipStream_t stream) {
  const int* x = (const int*)d_in[0];
  const float* emb = (const float*)d_in[1];
  const float* w_ih0 = (const float*)d_in[2];
  const float* w_ihr = (const float*)d_in[3];
  const float* w_hh = (const float*)d_in[4];
  const float* b_ih = (const float*)d_in[5];
  const float* b_hh = (const float*)d_in[6];
  const float* h2e_w = (const float*)d_in[7];
  const float* h2e_b = (const float*)d_in[8];
  const float* out_w = (const float*)d_in[9];
  const float* out_b = (const float*)d_in[10];

  char* ws = (char*)d_ws;
  u16* wi0b = (u16*)(ws + 0);          //  4*384*160 bf16
  u16* wirb = (u16*)(ws + 491520);     //  4*2*384*128
  u16* whhb = (u16*)(ws + 1277952);    //  4*3*384*128
  u16* hwb  = (u16*)(ws + 2457600);    //  4*160*128
  u16* owb  = (u16*)(ws + 2621440);    //  272*512
  u16* inpb = (u16*)(ws + 2899968);    //  4*32*32*32*160
  u16* combw= (u16*)(ws + 44843008);   //  4*32*32*160
  u16* prevh= (u16*)(ws + 46153728);   //  4*1024*128
  u16* o4h  = (u16*)(ws + 47202304);   //  32768*512
  float* hbp= (float*)(ws + 80756736); //  4*160 f32 (padded h2e_b)

  conv_kpad<<<960, 256, 0, stream>>>(wi0b, w_ih0, 4 * 384, 129, 160);
  conv_kpad<<<1536, 256, 0, stream>>>(wirb, w_ihr, 4 * 2 * 384, 128, 128);
  conv_kpad<<<2304, 256, 0, stream>>>(whhb, w_hh, 4 * 3 * 384, 128, 128);
  conv_rowpad<<<320, 256, 0, stream>>>(hwb, h2e_w, 4, 129, 160, 128);
  conv_rowpad<<<544, 256, 0, stream>>>(owb, out_w, 1, 258, 272, 512);
  pad_hb<<<3, 256, 0, stream>>>(hbp, h2e_b);
  embed_kernel<<<8192, 256, 0, stream>>>(inpb, x, emb);
  rnn_kernel<<<4, 512, 0, stream>>>(inpb, wi0b, wirb, whhb, hwb, b_ih, b_hh,
                                    hbp, combw, prevh, o4h);
  proj_kernel<<<dim3(2048, 3), 256, 0, stream>>>(o4h, owb, out_b, (float*)d_out);
}
